// RWKV7Attention_78709570667055
// MI455X (gfx1250) — compile-verified, matrix operand order checked by reading, not run on hardware
//
#include <hip/hip_runtime.h>

// ---------------------------------------------------------------------------
// RWKV-7 attention layer for MI455X (gfx1250), wave32 + WMMA bf16.
// Big GEMMs: v_wmma_f32_16x16x32_bf16, 128x64 tile / 256 threads,
// double-buffered LDS, A-tile staged via GLOBAL_LOAD_ASYNC_TO_LDS_B128
// (ASYNCcnt path) when the toolchain exposes the builtin.
// ---------------------------------------------------------------------------

typedef __bf16 bf16_t;
typedef __bf16 v16bf  __attribute__((ext_vector_type(16)));
typedef __bf16 bf16x8 __attribute__((ext_vector_type(8)));
typedef float  v8f    __attribute__((ext_vector_type(8)));
typedef int    v4i    __attribute__((ext_vector_type(4)));

#define GLOBAL_AS __attribute__((address_space(1)))
#define LDS_AS    __attribute__((address_space(3)))

#if defined(__HIP_DEVICE_COMPILE__) && \
    __has_builtin(__builtin_amdgcn_global_load_async_to_lds_b128)
#define ASYNC_LDS 1
#else
#define ASYNC_LDS 0
#endif

#if ASYNC_LDS
#if __has_builtin(__builtin_amdgcn_s_wait_asynccnt)
#define WAIT_ASYNC() __builtin_amdgcn_s_wait_asynccnt(0)
#else
#define WAIT_ASYNC() asm volatile("s_wait_asynccnt 0x0" ::: "memory")
#endif
#endif

#define C_DIM 2048
#define T_DIM 2048
#define B_DIM 4
#define H_DIM 32
#define N_DIM 64
#define BT    (B_DIM * T_DIM)          // 8192
#define BTC   ((long)BT * C_DIM)       // 16,777,216
#define LN_EPS 64e-5f

// ---------------------------------------------------------------------------
// WMMA GEMM:  C[M,N] (f32) = A[M,K] (bf16, row-major) * B[K,N] (bf16, row-major)
// Block: 256 threads = 8 waves in a 4(m) x 2(n) grid. Tile 128x64, K-step 32.
// Each wave computes a 32x32 patch = 4 accumulators -> 4 WMMA per K-step.
// Requires: M % 128 == 0, K % 32 == 0, N % 8 == 0.
// ---------------------------------------------------------------------------
__global__ __launch_bounds__(256) void gemm_bf16_wmma(
    const bf16_t* __restrict__ A, const bf16_t* __restrict__ Bw,
    float* __restrict__ Cc, int M, int N, int K) {
  __shared__ bf16_t As[2][128 * 32];   // [row][k], row stride 32 halves (64B)
  __shared__ bf16_t Bt[2][64 * 32];    // [n][k], transposed B tile

  const int tid  = threadIdx.x;
  const int m0   = blockIdx.y * 128;
  const int n0   = blockIdx.x * 64;
  const int wave = tid >> 5, lane = tid & 31;
  const int wm   = wave >> 1, wn = wave & 1;   // 4 x 2 wave grid
  const int hi   = lane >> 4, ml = lane & 15;

  v8f acc[2][2];
#pragma unroll
  for (int s = 0; s < 2; ++s)
#pragma unroll
    for (int t = 0; t < 2; ++t) acc[s][t] = v8f{};

  const int bk = tid >> 3;          // 0..31  (B tile k)
  const int bn = (tid & 7) * 8;     // 0..56  (B tile n chunk)

  auto loadB = [&](int k0, bf16x8& br) {
    if (n0 + bn + 8 <= N) {
      br = *(const bf16x8*)(Bw + (long)(k0 + bk) * N + n0 + bn);
    } else {
#pragma unroll
      for (int j = 0; j < 8; ++j) br[j] = (bf16_t)0.0f;   // N % 8 == 0
    }
  };
  auto stageB = [&](int buf, bf16x8 br) {
#pragma unroll
    for (int j = 0; j < 8; ++j) Bt[buf][(bn + j) * 32 + bk] = br[j];
  };
  // per-lane fragment: two contiguous 16B LDS reads -> ds_load_b128 pair
  auto frag = [&](const bf16_t* row) {
    bf16x8 lo = *(const bf16x8*)(row + hi * 8);
    bf16x8 hb = *(const bf16x8*)(row + 16 + hi * 8);
    return __builtin_shufflevector(lo, hb, 0, 1, 2, 3, 4, 5, 6, 7,
                                   8, 9, 10, 11, 12, 13, 14, 15);
  };

#if ASYNC_LDS
  // --- A tile staged directly memory -> LDS via async DMA (ASYNCcnt path)
  auto asyncA = [&](int k0, int buf) {
#pragma unroll
    for (int i = 0; i < 2; ++i) {
      int idx = tid + i * 256;            // 0..511
      int row = idx >> 2;                 // 0..127
      int col = (idx & 3) * 8;            // 0,8,16,24
      __builtin_amdgcn_global_load_async_to_lds_b128(
          (GLOBAL_AS v4i*)(A + (long)(m0 + row) * K + k0 + col),
          (LDS_AS v4i*)&As[buf][row * 32 + col], 0, 0);
    }
  };

  {
    bf16x8 b0;
    asyncA(0, 0);
    loadB(0, b0); stageB(0, b0);
    WAIT_ASYNC();
    __syncthreads();
  }

  int buf = 0;
  for (int k0 = 0; k0 < K; k0 += 32) {
    const bool last = (k0 + 32 >= K);
    if (!last) {
      // stage next tile into buf^1 (safe: all reads of buf^1 completed
      // before the barrier that ended the previous iteration)
      asyncA(k0 + 32, buf ^ 1);
      bf16x8 bR;
      loadB(k0 + 32, bR); stageB(buf ^ 1, bR);
      if (k0 + 64 < K) {   // global_prefetch_b8 for tile k+2
        __builtin_prefetch(A + (long)(m0 + (tid >> 2)) * K + k0 + 64, 0, 1);
        __builtin_prefetch(Bw + (long)(k0 + 64 + bk) * N + n0, 0, 1);
      }
    }

    v16bf af0 = frag(&As[buf][(wm * 32 + ml) * 32]);
    v16bf af1 = frag(&As[buf][(wm * 32 + 16 + ml) * 32]);
    v16bf bf0 = frag(&Bt[buf][(wn * 32 + ml) * 32]);
    v16bf bf1 = frag(&Bt[buf][(wn * 32 + 16 + ml) * 32]);

    acc[0][0] = __builtin_amdgcn_wmma_f32_16x16x32_bf16(false, af0, false, bf0,
                                                        (short)0, acc[0][0], false, false);
    acc[0][1] = __builtin_amdgcn_wmma_f32_16x16x32_bf16(false, af0, false, bf1,
                                                        (short)0, acc[0][1], false, false);
    acc[1][0] = __builtin_amdgcn_wmma_f32_16x16x32_bf16(false, af1, false, bf0,
                                                        (short)0, acc[1][0], false, false);
    acc[1][1] = __builtin_amdgcn_wmma_f32_16x16x32_bf16(false, af1, false, bf1,
                                                        (short)0, acc[1][1], false, false);
    if (!last) {
      WAIT_ASYNC();        // own async stores to buf^1 landed
      __syncthreads();     // everyone's buf^1 writes visible, buf reads done
      buf ^= 1;
    }
  }
#else
  // --- fallback: register staging, two barriers per K-step
  auto loadA = [&](int k0, bf16x8* ar) {
#pragma unroll
    for (int i = 0; i < 2; ++i) {
      int idx = tid + i * 256;
      int row = idx >> 2;
      int col = (idx & 3) * 8;
      ar[i] = *(const bf16x8*)(A + (long)(m0 + row) * K + k0 + col);
    }
  };
  auto stageA = [&](int buf, const bf16x8* ar) {
#pragma unroll
    for (int i = 0; i < 2; ++i) {
      int idx = tid + i * 256;
      int row = idx >> 2;
      int col = (idx & 3) * 8;
      *(bf16x8*)&As[buf][row * 32 + col] = ar[i];
    }
  };

  bf16x8 aR[2]; bf16x8 bR;
  loadA(0, aR); loadB(0, bR);
  stageA(0, aR); stageB(0, bR);
  __syncthreads();

  int buf = 0;
  for (int k0 = 0; k0 < K; k0 += 32) {
    const bool last = (k0 + 32 >= K);
    if (!last) {
      loadA(k0 + 32, aR);
      loadB(k0 + 32, bR);
      if (k0 + 64 < K) {
        __builtin_prefetch(A + (long)(m0 + (tid >> 2)) * K + k0 + 64, 0, 1);
        __builtin_prefetch(Bw + (long)(k0 + 64 + bk) * N + n0, 0, 1);
      }
    }

    v16bf af0 = frag(&As[buf][(wm * 32 + ml) * 32]);
    v16bf af1 = frag(&As[buf][(wm * 32 + 16 + ml) * 32]);
    v16bf bf0 = frag(&Bt[buf][(wn * 32 + ml) * 32]);
    v16bf bf1 = frag(&Bt[buf][(wn * 32 + 16 + ml) * 32]);

    acc[0][0] = __builtin_amdgcn_wmma_f32_16x16x32_bf16(false, af0, false, bf0,
                                                        (short)0, acc[0][0], false, false);
    acc[0][1] = __builtin_amdgcn_wmma_f32_16x16x32_bf16(false, af0, false, bf1,
                                                        (short)0, acc[0][1], false, false);
    acc[1][0] = __builtin_amdgcn_wmma_f32_16x16x32_bf16(false, af1, false, bf0,
                                                        (short)0, acc[1][0], false, false);
    acc[1][1] = __builtin_amdgcn_wmma_f32_16x16x32_bf16(false, af1, false, bf1,
                                                        (short)0, acc[1][1], false, false);
    __syncthreads();
    if (!last) {
      stageA(buf ^ 1, aR);
      stageB(buf ^ 1, bR);
      __syncthreads();
      buf ^= 1;
    }
  }
#endif

  // store: lane holds column (wn*32 + t*16 + ml), rows g + 8*hi per accumulator
#pragma unroll
  for (int t = 0; t < 2; ++t) {
    int ncol = n0 + wn * 32 + t * 16 + ml;
    if (ncol < N) {
      float* cp = Cc + (long)(m0 + wm * 32 + hi * 8) * N + ncol;
#pragma unroll
      for (int s = 0; s < 2; ++s) {
        float* cps = cp + (long)s * 16 * N;
#pragma unroll
        for (int g = 0; g < 8; ++g) {
          *cps = acc[s][t][g];
          cps += N;
        }
      }
    }
  }
}

// ---------------------------------------------------------------------------
// Token-shift mix:  out_bf16 = x + (x_prev - x) * m[c]   (x_prev = 0 at t=0)
// ---------------------------------------------------------------------------
__global__ void mix_kernel(const float* __restrict__ x,
                           const float* __restrict__ m,
                           bf16_t* __restrict__ out) {
  long i = (long)blockIdx.x * blockDim.x + threadIdx.x;
  int  c = (int)(i & (C_DIM - 1));
  long t = (i >> 11) & (T_DIM - 1);
  float xv = x[i];
  float xp = (t == 0) ? 0.0f : x[i - C_DIM];
  out[i] = (bf16_t)(xv + (xp - xv) * m[c]);
}

// f32 -> bf16 with optional activation. mode: 0 copy, 1 tanh, 2 sigmoid.
__global__ void cvt_act(const float* __restrict__ in, bf16_t* __restrict__ out,
                        long n, int mode) {
  long i = (long)blockIdx.x * blockDim.x + threadIdx.x;
  if (i >= n) return;
  float v = in[i];
  if (mode == 1) v = tanhf(v);
  else if (mode == 2) v = 1.0f / (1.0f + expf(-v));
  out[i] = (bf16_t)v;
}

// in-place:  x = sigmoid(x + bias[c])
__global__ void sigmoid_bias_inplace(float* __restrict__ x,
                                     const float* __restrict__ bias) {
  long i = (long)blockIdx.x * blockDim.x + threadIdx.x;
  float v = x[i] + bias[i & (C_DIM - 1)];
  x[i] = 1.0f / (1.0f + expf(-v));
}

// in-place decay: w = exp(-exp(-softplus(-(raw + w0)) - 0.5))
__global__ void decay_inplace(float* __restrict__ x,
                              const float* __restrict__ w0) {
  long i = (long)blockIdx.x * blockDim.x + threadIdx.x;
  float raw = x[i] + w0[i & (C_DIM - 1)];
  float z  = -raw;
  float sp = (z > 20.0f) ? z : log1pf(expf(z));   // softplus(-raw)
  x[i] = expf(-expf(-sp - 0.5f));
}

// value-residual mix:  v = v + (v_first - v) * sigmoid(raw + v0[c])
__global__ void vmix_kernel(float* __restrict__ v,
                            const float* __restrict__ vf,
                            const float* __restrict__ raw,
                            const float* __restrict__ v0) {
  long i = (long)blockIdx.x * blockDim.x + threadIdx.x;
  float s = 1.0f / (1.0f + expf(-(raw[i] + v0[i & (C_DIM - 1)])));
  float vv = v[i];
  v[i] = vv + (vf[i] - vv) * s;
}

// ---------------------------------------------------------------------------
// kk normalize + k/kka fuse. One 64-thread block per (b,t,h).
// ---------------------------------------------------------------------------
__global__ __launch_bounds__(64) void kk_fuse(
    float* __restrict__ k, const float* __restrict__ a,
    const float* __restrict__ k_k, const float* __restrict__ k_a,
    float* __restrict__ kk, float* __restrict__ kka) {
  long bh = blockIdx.x;
  int tid = threadIdx.x;
  long i = bh * 64 + tid;
  int  c = (int)((bh % H_DIM) * 64 + tid);
  __shared__ float red[64];

  float kv  = k[i];
  float kkv = kv * k_k[c];
  red[tid] = kkv * kkv;
  __syncthreads();
#pragma unroll
  for (int off = 32; off > 0; off >>= 1) {
    if (tid < off) red[tid] += red[tid + off];
    __syncthreads();
  }
  float den = fmaxf(sqrtf(red[0]), 1e-12f);
  float kkn = kkv / den;
  float av  = a[i];
  kk[i]  = kkn;
  kka[i] = kkn * av;
  k[i]   = kv * (1.0f + (av - 1.0f) * k_a[c]);
}

// ---------------------------------------------------------------------------
// WKV-7 scan. One 64-thread block per (b,h); thread i owns state row i.
// ---------------------------------------------------------------------------
__global__ __launch_bounds__(64) void wkv_scan(
    const float* __restrict__ r, const float* __restrict__ w,
    const float* __restrict__ k, const float* __restrict__ v,
    const float* __restrict__ kk, const float* __restrict__ kka,
    float* __restrict__ y) {
  const int b = blockIdx.x / H_DIM;
  const int h = blockIdx.x % H_DIM;
  const int tid = threadIdx.x;
  const long base = (long)b * T_DIM * C_DIM + h * N_DIM;

  __shared__ float rs[64], ws[64], ks[64], kks[64], kkas[64], vs[64];
  float s[64];
#pragma unroll
  for (int j = 0; j < 64; ++j) s[j] = 0.0f;

  for (int t = 0; t < T_DIM; ++t) {
    long o = base + (long)t * C_DIM;
    rs[tid] = r[o + tid];  ws[tid] = w[o + tid];  ks[tid] = k[o + tid];
    kks[tid] = kk[o + tid]; kkas[tid] = kka[o + tid]; vs[tid] = v[o + tid];
    __syncthreads();

    float sa = 0.0f;
#pragma unroll
    for (int j = 0; j < 64; ++j) sa -= s[j] * kks[j];

    float vi = vs[tid];
    float yi = 0.0f;
#pragma unroll
    for (int j = 0; j < 64; ++j) {
      s[j] = s[j] * ws[j] + sa * kkas[j] + vi * ks[j];
      yi += s[j] * rs[j];
    }
    y[o + tid] = yi;
    __syncthreads();
  }
}

// ---------------------------------------------------------------------------
// Per-head group norm + bonus + gate.  One 64-thread block per (b,t,h).
// ---------------------------------------------------------------------------
__global__ __launch_bounds__(64) void post_kernel(
    const float* __restrict__ y, const float* __restrict__ r,
    const float* __restrict__ k, const float* __restrict__ v,
    const float* __restrict__ g, const float* __restrict__ r_k,
    const float* __restrict__ lnw, const float* __restrict__ lnb,
    bf16_t* __restrict__ og) {
  long bh = blockIdx.x;
  int tid = threadIdx.x;
  long i = bh * 64 + tid;
  int  c = (int)((bh % H_DIM) * 64 + tid);
  __shared__ float red[64];

  float yv = y[i];
  red[tid] = yv; __syncthreads();
#pragma unroll
  for (int off = 32; off > 0; off >>= 1) {
    if (tid < off) red[tid] += red[tid + off];
    __syncthreads();
  }
  float mu = red[0] * (1.0f / 64.0f);
  __syncthreads();
  float dy = yv - mu;
  red[tid] = dy * dy; __syncthreads();
#pragma unroll
  for (int off = 32; off > 0; off >>= 1) {
    if (tid < off) red[tid] += red[tid + off];
    __syncthreads();
  }
  float var = red[0] * (1.0f / 64.0f);
  __syncthreads();
  red[tid] = r[i] * k[i] * r_k[c]; __syncthreads();
#pragma unroll
  for (int off = 32; off > 0; off >>= 1) {
    if (tid < off) red[tid] += red[tid + off];
    __syncthreads();
  }
  float dot = red[0];

  float yn = dy * rsqrtf(var + LN_EPS) * lnw[c] + lnb[c];
  og[i] = (bf16_t)((yn + dot * v[i]) * g[i]);
}

// ---------------------------------------------------------------------------
// Host-side launch
// ---------------------------------------------------------------------------
extern "C" void kernel_launch(void* const* d_in, const int* in_sizes, int n_in,
                              void* d_out, int out_size, void* d_ws, size_t ws_size,
                              hipStream_t stream) {
  const float* x      = (const float*)d_in[0];
  const float* vfirst = (const float*)d_in[1];
  const float* x_r = (const float*)d_in[2];
  const float* x_w = (const float*)d_in[3];
  const float* x_k = (const float*)d_in[4];
  const float* x_v = (const float*)d_in[5];
  const float* x_a = (const float*)d_in[6];
  const float* x_g = (const float*)d_in[7];
  const float* w0 = (const float*)d_in[8];
  const float* w1 = (const float*)d_in[9];
  const float* w2 = (const float*)d_in[10];
  const float* a0 = (const float*)d_in[11];
  const float* a1 = (const float*)d_in[12];
  const float* a2 = (const float*)d_in[13];
  const float* v0 = (const float*)d_in[14];
  const float* v1 = (const float*)d_in[15];
  const float* v2 = (const float*)d_in[16];
  const float* g1 = (const float*)d_in[17];
  const float* g2 = (const float*)d_in[18];
  const float* k_k = (const float*)d_in[19];
  const float* k_a = (const float*)d_in[20];
  const float* r_k = (const float*)d_in[21];
  const float* W_r = (const float*)d_in[22];
  const float* W_k = (const float*)d_in[23];
  const float* W_v = (const float*)d_in[24];
  const float* W_o = (const float*)d_in[25];
  const float* lnw = (const float*)d_in[26];
  const float* lnb = (const float*)d_in[27];
  float* out = (float*)d_out;

  // ---- workspace carving
  char* base = (char*)d_ws;
  size_t off = 0;
  auto carve = [&](size_t bytes) {
    char* p = base + off;
    off += (bytes + 255) & ~(size_t)255;
    return (void*)p;
  };
  const size_t CC = (size_t)C_DIM * C_DIM;
  bf16_t* Wr_bf = (bf16_t*)carve(CC * 2);
  bf16_t* Wk_bf = (bf16_t*)carve(CC * 2);
  bf16_t* Wv_bf = (bf16_t*)carve(CC * 2);
  bf16_t* Wo_bf = (bf16_t*)carve(CC * 2);
  bf16_t* w1_bf = (bf16_t*)carve((size_t)C_DIM * 96 * 2);
  bf16_t* w2_bf = (bf16_t*)carve((size_t)96 * C_DIM * 2);
  bf16_t* a1_bf = (bf16_t*)carve((size_t)C_DIM * 96 * 2);
  bf16_t* a2_bf = (bf16_t*)carve((size_t)96 * C_DIM * 2);
  bf16_t* v1_bf = (bf16_t*)carve((size_t)C_DIM * 64 * 2);
  bf16_t* v2_bf = (bf16_t*)carve((size_t)64 * C_DIM * 2);
  bf16_t* g1_bf = (bf16_t*)carve((size_t)C_DIM * 256 * 2);
  bf16_t* g2_bf = (bf16_t*)carve((size_t)256 * C_DIM * 2);
  bf16_t* xbf   = (bf16_t*)carve((size_t)BTC * 2);
  bf16_t* ogbf  = (bf16_t*)carve((size_t)BTC * 2);
  bf16_t* midbf = (bf16_t*)carve((size_t)BT * 256 * 2);
  float*  midf  = (float*)carve((size_t)BT * 256 * 4);
  float*  rbuf  = (float*)carve((size_t)BTC * 4);
  float*  kbuf  = (float*)carve((size_t)BTC * 4);
  float*  vbuf  = (float*)carve((size_t)BTC * 4);
  float*  kkbuf = (float*)carve((size_t)BTC * 4);
  float*  kkabuf= (float*)carve((size_t)BTC * 4);
  float*  tmp   = (float*)carve((size_t)BTC * 4);   // a / w_raw / decay
  float*  gbuf  = (float*)carve((size_t)BTC * 4);
  float*  ybuf  = (float*)carve((size_t)BTC * 4);
  (void)ws_size; (void)in_sizes; (void)n_in; (void)out_size;

  const int EWB = 256;
  auto ew_grid = [&](long n) { return dim3((unsigned)((n + EWB - 1) / EWB)); };
  auto gemm_grid = [&](int M, int N) { return dim3((unsigned)((N + 63) / 64), (unsigned)(M / 128)); };

  // ---- weight conversion to bf16
  cvt_act<<<ew_grid(CC), EWB, 0, stream>>>(W_r, Wr_bf, (long)CC, 0);
  cvt_act<<<ew_grid(CC), EWB, 0, stream>>>(W_k, Wk_bf, (long)CC, 0);
  cvt_act<<<ew_grid(CC), EWB, 0, stream>>>(W_v, Wv_bf, (long)CC, 0);
  cvt_act<<<ew_grid(CC), EWB, 0, stream>>>(W_o, Wo_bf, (long)CC, 0);
  cvt_act<<<ew_grid(C_DIM * 96), EWB, 0, stream>>>(w1, w1_bf, (long)C_DIM * 96, 0);
  cvt_act<<<ew_grid(96 * C_DIM), EWB, 0, stream>>>(w2, w2_bf, (long)96 * C_DIM, 0);
  cvt_act<<<ew_grid(C_DIM * 96), EWB, 0, stream>>>(a1, a1_bf, (long)C_DIM * 96, 0);
  cvt_act<<<ew_grid(96 * C_DIM), EWB, 0, stream>>>(a2, a2_bf, (long)96 * C_DIM, 0);
  cvt_act<<<ew_grid(C_DIM * 64), EWB, 0, stream>>>(v1, v1_bf, (long)C_DIM * 64, 0);
  cvt_act<<<ew_grid(64 * C_DIM), EWB, 0, stream>>>(v2, v2_bf, (long)64 * C_DIM, 0);
  cvt_act<<<ew_grid(C_DIM * 256), EWB, 0, stream>>>(g1, g1_bf, (long)C_DIM * 256, 0);
  cvt_act<<<ew_grid(256 * C_DIM), EWB, 0, stream>>>(g2, g2_bf, (long)256 * C_DIM, 0);

  // ---- r = xr @ W_r
  mix_kernel<<<ew_grid(BTC), EWB, 0, stream>>>(x, x_r, xbf);
  gemm_bf16_wmma<<<gemm_grid(BT, C_DIM), 256, 0, stream>>>(xbf, Wr_bf, rbuf, BT, C_DIM, C_DIM);
  // ---- k = xk @ W_k
  mix_kernel<<<ew_grid(BTC), EWB, 0, stream>>>(x, x_k, xbf);
  gemm_bf16_wmma<<<gemm_grid(BT, C_DIM), 256, 0, stream>>>(xbf, Wk_bf, kbuf, BT, C_DIM, C_DIM);
  // ---- v = xv @ W_v, then value-residual mix via v-lora
  mix_kernel<<<ew_grid(BTC), EWB, 0, stream>>>(x, x_v, xbf);
  gemm_bf16_wmma<<<gemm_grid(BT, C_DIM), 256, 0, stream>>>(xbf, Wv_bf, vbuf, BT, C_DIM, C_DIM);
  gemm_bf16_wmma<<<gemm_grid(BT, 64), 256, 0, stream>>>(xbf, v1_bf, midf, BT, 64, C_DIM);
  cvt_act<<<ew_grid((long)BT * 64), EWB, 0, stream>>>(midf, midbf, (long)BT * 64, 0);
  gemm_bf16_wmma<<<gemm_grid(BT, C_DIM), 256, 0, stream>>>(midbf, v2_bf, tmp, BT, C_DIM, 64);
  vmix_kernel<<<ew_grid(BTC), EWB, 0, stream>>>(vbuf, vfirst, tmp, v0);
  // ---- a = sigmoid(xa@a1@a2 + a0)   (a lives in tmp)
  mix_kernel<<<ew_grid(BTC), EWB, 0, stream>>>(x, x_a, xbf);
  gemm_bf16_wmma<<<gemm_grid(BT, 96), 256, 0, stream>>>(xbf, a1_bf, midf, BT, 96, C_DIM);
  cvt_act<<<ew_grid((long)BT * 96), EWB, 0, stream>>>(midf, midbf, (long)BT * 96, 0);
  gemm_bf16_wmma<<<gemm_grid(BT, C_DIM), 256, 0, stream>>>(midbf, a2_bf, tmp, BT, C_DIM, 96);
  sigmoid_bias_inplace<<<ew_grid(BTC), EWB, 0, stream>>>(tmp, a0);
  // ---- kk normalize, kka, k update (consumes a; frees tmp)
  kk_fuse<<<dim3((unsigned)(BT * H_DIM)), 64, 0, stream>>>(kbuf, tmp, k_k, k_a, kkbuf, kkabuf);
  // ---- decay w (into tmp)
  mix_kernel<<<ew_grid(BTC), EWB, 0, stream>>>(x, x_w, xbf);
  gemm_bf16_wmma<<<gemm_grid(BT, 96), 256, 0, stream>>>(xbf, w1_bf, midf, BT, 96, C_DIM);
  cvt_act<<<ew_grid((long)BT * 96), EWB, 0, stream>>>(midf, midbf, (long)BT * 96, 1);  // tanh
  gemm_bf16_wmma<<<gemm_grid(BT, C_DIM), 256, 0, stream>>>(midbf, w2_bf, tmp, BT, C_DIM, 96);
  decay_inplace<<<ew_grid(BTC), EWB, 0, stream>>>(tmp, w0);
  // ---- gate g
  mix_kernel<<<ew_grid(BTC), EWB, 0, stream>>>(x, x_g, xbf);
  gemm_bf16_wmma<<<gemm_grid(BT, 256), 256, 0, stream>>>(xbf, g1_bf, midf, BT, 256, C_DIM);
  cvt_act<<<ew_grid((long)BT * 256), EWB, 0, stream>>>(midf, midbf, (long)BT * 256, 2); // sigmoid
  gemm_bf16_wmma<<<gemm_grid(BT, C_DIM), 256, 0, stream>>>(midbf, g2_bf, gbuf, BT, C_DIM, 256);
  // ---- sequential WKV scan
  wkv_scan<<<dim3(B_DIM * H_DIM), 64, 0, stream>>>(rbuf, tmp, kbuf, vbuf, kkbuf, kkabuf, ybuf);
  // ---- groupnorm + bonus + gate -> bf16
  post_kernel<<<dim3((unsigned)(BT * H_DIM)), 64, 0, stream>>>(
      ybuf, rbuf, kbuf, vbuf, gbuf, r_k, lnw, lnb, ogbf);
  // ---- final projection
  gemm_bf16_wmma<<<gemm_grid(BT, C_DIM), 256, 0, stream>>>(ogbf, Wo_bf, out, BT, C_DIM, C_DIM);
}
